// PoseGatEncoder_68247030333771
// MI455X (gfx1250) — matrix operand order repaired and checked by reading, not compiled, not run on hardware
//
#include <hip/hip_runtime.h>
#include <hip/hip_bf16.h>

// ---------------------------------------------------------------------------
// PoseGatEncoder for MI455X (gfx1250, wave32, WMMA)
//   proj (WMMA f16->f32)  ->  chain-graph GATv2 attention (VALU, degree<=3)
//   -> ELU -> ... -> output head GEMM (WMMA + TDM tensor_load_to_lds)
// ---------------------------------------------------------------------------

typedef _Float16 h16 __attribute__((ext_vector_type(16)));
typedef _Float16 h8  __attribute__((ext_vector_type(8)));
typedef float    f8  __attribute__((ext_vector_type(8)));
typedef unsigned int u32x4 __attribute__((ext_vector_type(4)));
typedef int          i32x8 __attribute__((ext_vector_type(8)));
typedef int          i32x4 __attribute__((ext_vector_type(4)));

#if defined(__HIP_DEVICE_COMPILE__) && __has_builtin(__builtin_amdgcn_tensor_load_to_lds)
#define HAVE_TDM 1
#else
#define HAVE_TDM 0
#endif

#define NJ   54
#define NG   3072            // B*T
#define MROW (NG * NJ)       // 165888 node rows
#define KOUT 13824           // NJ*256
#define NOUTF 512

// ---- WMMA fragment helpers (ISA 7.12.2 layouts, wave32) -------------------
// A 16x32 f16: lane&15 = M row; lane>>4 selects K-halves {0..7,16..23} vs {8..15,24..31}
__device__ inline h16 load_a_frag(const _Float16* t, int ld) {
  const int lane = threadIdx.x & 31;
  const _Float16* r = t + (lane & 15) * ld + ((lane >> 4) << 3);
  h8 lo = *(const h8*)r;
  h8 hi = *(const h8*)(r + 16);
  return __builtin_shufflevector(lo, hi, 0,1,2,3,4,5,6,7,8,9,10,11,12,13,14,15);
}
// B 32x16 f16, staged N-major in LDS (bt[n*ldk + k]): lane>>4 selects K-halves {0..7,16..23} vs {8..15,24..31}
__device__ inline h16 load_b_frag(const _Float16* bt, int ldk) {
  const int lane = threadIdx.x & 31;
  const _Float16* r = bt + (lane & 15) * ldk + ((lane >> 4) << 3);
  h8 lo = *(const h8*)r;
  h8 hi = *(const h8*)(r + 16);
  return __builtin_shufflevector(lo, hi, 0,1,2,3,4,5,6,7,8,9,10,11,12,13,14,15);
}

__device__ inline f8 wmma_f16(h16 a, h16 b, f8 c) {
  return __builtin_amdgcn_wmma_f32_16x16x32_f16(false, a, false, b, (short)0, c,
                                                false, false);
}

// ---- conversion / transposition kernels -----------------------------------
__global__ __launch_bounds__(256) void cvt_f32_to_f16_kernel(
    const float* __restrict__ in, _Float16* __restrict__ out, int n) {
  int i = blockIdx.x * 256 + threadIdx.x;
  if (i < n) out[i] = (_Float16)in[i];
}

// W (din x nout, row-major f32) -> Wt (nout x K32, row-major f16), zero-pad K
__global__ __launch_bounds__(256) void cvt_w_transpose_kernel(
    const float* __restrict__ W, _Float16* __restrict__ Wt,
    int din, int nout, int K32) {
  int i = blockIdx.x * 256 + threadIdx.x;
  if (i >= nout * K32) return;
  int n = i / K32, k = i - n * K32;
  Wt[i] = (k < din) ? (_Float16)W[(size_t)k * nout + n] : (_Float16)0.f;
}

// ---- projection GEMM: hl = x@W_l+b_l (z=0), hr = x@W_r+b_r (z=1) ----------
// A: MROW x din (f16), Bt: nout x K32 (f16, N-major), out: MROW x nout (f16)
template <int K32>
__global__ __launch_bounds__(256) void proj_wmma_kernel(
    const _Float16* __restrict__ A, int din, int nout,
    const _Float16* __restrict__ BtL, const _Float16* __restrict__ BtR,
    const float* __restrict__ bL, const float* __restrict__ bR,
    _Float16* __restrict__ outL, _Float16* __restrict__ outR) {
  __shared__ __align__(64) _Float16 Atile[128 * K32];
  __shared__ __align__(64) _Float16 Btile[64 * K32];
  const _Float16* Bt  = blockIdx.z ? BtR : BtL;
  const float*   bias = blockIdx.z ? bR  : bL;
  _Float16*      out  = blockIdx.z ? outR : outL;
  const int mbase = blockIdx.x * 128;
  const int nbase = blockIdx.y * 64;
  const int tid = threadIdx.x;

  // stage A into LDS (b128 fast path when rows are unpadded/contiguous)
  if (din == K32) {
    const uint4* src = (const uint4*)(A + (size_t)mbase * K32);
    uint4* dst = (uint4*)Atile;
    for (int i = tid; i < 128 * K32 / 8; i += 256) dst[i] = src[i];
  } else {
    for (int i = tid; i < 128 * K32; i += 256) {
      int r = i / K32, k = i - r * K32;
      Atile[i] = (k < din) ? A[(size_t)(mbase + r) * din + k] : (_Float16)0.f;
    }
  }
  // stage Bt rows [nbase, nbase+64) — contiguous in global, b128 copies
  {
    const uint4* src = (const uint4*)(Bt + (size_t)nbase * K32);
    uint4* dst = (uint4*)Btile;
    for (int i = tid; i < 64 * K32 / 8; i += 256) dst[i] = src[i];
  }
  __syncthreads();

  const int w = tid >> 5, lane = tid & 31;
  h16 af[K32 / 32];
#pragma unroll
  for (int kt = 0; kt < K32 / 32; ++kt)
    af[kt] = load_a_frag(Atile + w * 16 * K32 + kt * 32, K32);

  const int col = lane & 15;
  const int rofs = (lane >> 4) << 3;
#pragma unroll
  for (int nt = 0; nt < 4; ++nt) {
    f8 acc = {};
#pragma unroll
    for (int kt = 0; kt < K32 / 32; ++kt) {
      h16 bf = load_b_frag(Btile + (nt * 16) * K32 + kt * 32, K32);
      acc = wmma_f16(af[kt], bf, acc);
    }
    const int gc = nbase + nt * 16 + col;
    const float bv = bias[gc];
    const size_t row0 = (size_t)mbase + w * 16 + rofs;
#pragma unroll
    for (int r = 0; r < 8; ++r)
      out[(row0 + r) * nout + gc] = (_Float16)(acc[r] + bv);
  }
}

// ---- GATv2 attention over the chain graph (in-degree <= 3) ----------------
// one thread per (graph, node, head); H = 4 fixed
__global__ __launch_bounds__(256) void gat_attn_kernel(
    const _Float16* __restrict__ hl, const _Float16* __restrict__ hr,
    const float* __restrict__ att, const float* __restrict__ bias,
    _Float16* __restrict__ out, int C) {
  const int t = blockIdx.x * 256 + threadIdx.x;
  const int h = t & 3;
  const int j = (t >> 2) % NJ;
  const int g = (t >> 2) / NJ;
  const int HC = 4 * C;
  const size_t node = (size_t)g * NJ + j;
  const _Float16* hrj = hr + node * HC + h * C;
  const float* av = att + h * C;

  int srcs[3]; int ns = 0;
  if (j > 0)      srcs[ns++] = j - 1;
  if (j < NJ - 1) srcs[ns++] = j + 1;
  srcs[ns++] = j;  // self edge

  float sc[3];
  for (int k = 0; k < ns; ++k) {
    const _Float16* hls = hl + ((size_t)g * NJ + srcs[k]) * HC + h * C;
    float s = 0.f;
    for (int c = 0; c < C; ++c) {
      float e = (float)hls[c] + (float)hrj[c];
      e = e > 0.f ? e : 0.2f * e;          // leaky_relu 0.2
      s += e * av[c];
    }
    sc[k] = s;
  }
  float m = sc[0];
  for (int k = 1; k < ns; ++k) m = fmaxf(m, sc[k]);
  float ex[3], den = 0.f;
  for (int k = 0; k < ns; ++k) { ex[k] = __expf(sc[k] - m); den += ex[k]; }
  const float inv = 1.f / den;

  _Float16* o = out + node * HC + h * C;
  const float* bb = bias + h * C;
  for (int c = 0; c < C; ++c) {
    float acc = 0.f;
    for (int k = 0; k < ns; ++k)
      acc += ex[k] * inv *
             (float)hl[((size_t)g * NJ + srcs[k]) * HC + h * C + c];
    acc += bb[c];
    acc = acc > 0.f ? acc : expm1f(acc);   // ELU
    o[c] = (_Float16)acc;
  }
}

// ---- output head GEMM: (3072 x 13824) @ Wt^T (+bias) via WMMA + TDM -------
// A: NG x KOUT f16; Bt: NOUTF x KOUT f16 (N-major); out: NG x NOUTF f32
__global__ __launch_bounds__(256) void final_gemm_kernel(
    const _Float16* __restrict__ A, const _Float16* __restrict__ Bt,
    const float* __restrict__ bias, float* __restrict__ out) {
  __shared__ __align__(64) _Float16 Atile[128 * 32];
  __shared__ __align__(64) _Float16 Btile[128 * 32];  // N-major: [n][k]
  const int mbase = blockIdx.x * 128, nbase = blockIdx.y * 128;
  const int tid = threadIdx.x, w = tid >> 5, lane = tid & 31;

  f8 acc[8];
#pragma unroll
  for (int i = 0; i < 8; ++i) acc[i] = (f8){};

#if HAVE_TDM
  // Tensor DMA descriptor (D#) pieces — 2D tile 32(K) x 128(N) of f16
  const unsigned ldsb = (unsigned)(size_t)(void*)Btile;      // flat addr[31:0] = LDS offset
  const unsigned long long gbase =
      (unsigned long long)(size_t)(const void*)Bt +
      (unsigned long long)nbase * KOUT * 2ull;
  i32x8 g1;
  g1[0] = 1 << 16;                       // data_size = 2 bytes
  g1[1] = (KOUT & 0xFFFF) << 16;         // tensor_dim0 = 13824 (lo16)
  g1[2] = (KOUT >> 16) | (NOUTF << 16);  // tensor_dim0 hi16 | tensor_dim1 lo16
  g1[3] = (NOUTF >> 16) | (32 << 16);    // tensor_dim1 hi16 | tile_dim0 = 32
  g1[4] = 128;                           // tile_dim1 = 128
  g1[5] = KOUT;                          // tensor_dim0_stride lo32
  g1[6] = 0; g1[7] = 0;
  const i32x4 gz  = {0, 0, 0, 0};
  const i32x8 gz8 = {0, 0, 0, 0, 0, 0, 0, 0};
#endif

  for (int kt = 0; kt < KOUT / 32; ++kt) {
    const int k0 = kt * 32;
    // stage A panel (128 rows x 32 halves) as b128 vectors: 2 per thread
    {
      const uint4* src = (const uint4*)(A + (size_t)mbase * KOUT + k0);
      uint4* dst = (uint4*)Atile;
#pragma unroll
      for (int i = tid; i < 512; i += 256) {
        const int r = i >> 2, q = i & 3;  // 4 x b128 per 32-half row
        dst[i] = src[(size_t)r * (KOUT / 8) + q];
      }
    }
    // warm next A panel in L2 (global_prefetch_b8)
    if (kt + 1 < KOUT / 32 && tid < 128)
      __builtin_prefetch(
          (const void*)(A + (size_t)(mbase + tid) * KOUT + k0 + 32), 0, 1);

#if HAVE_TDM
    if (tid < 32) {  // wave 0 issues the tensor DMA for the B panel
      const unsigned long long ga = gbase + (unsigned long long)k0 * 2ull;
      u32x4 g0;
      g0[0] = 1u;                                  // count=1, user mode
      g0[1] = ldsb;                                // lds_addr
      g0[2] = (unsigned)ga;                        // global_addr lo
      g0[3] = (unsigned)(ga >> 32) | (2u << 30);   // addr hi | type=2
      __builtin_amdgcn_tensor_load_to_lds(g0, g1, gz, gz, gz8, 0);
      __builtin_amdgcn_s_wait_tensorcnt(0);
    }
#else
    {
      const uint4* src = (const uint4*)(Bt + (size_t)nbase * KOUT + k0);
      uint4* dst = (uint4*)Btile;
#pragma unroll
      for (int i = tid; i < 512; i += 256) {
        const int n = i >> 2, q = i & 3;
        dst[i] = src[(size_t)n * (KOUT / 8) + q];
      }
    }
#endif
    __syncthreads();

    h16 af = load_a_frag(Atile + w * 16 * 32, 32);
#pragma unroll
    for (int nt = 0; nt < 8; ++nt) {
      h16 bf = load_b_frag(Btile + nt * 16 * 32, 32);
      acc[nt] = wmma_f16(af, bf, acc[nt]);
    }
    __syncthreads();
  }

  const int col0 = nbase + (lane & 15);
  const int rbase = mbase + w * 16 + ((lane >> 4) << 3);
#pragma unroll
  for (int nt = 0; nt < 8; ++nt) {
    const int c = col0 + nt * 16;
    const float bv = bias[c];
#pragma unroll
    for (int r = 0; r < 8; ++r)
      out[(size_t)(rbase + r) * NOUTF + c] = acc[nt][r] + bv;
  }
}

// ---------------------------------------------------------------------------
extern "C" void kernel_launch(void* const* d_in, const int* in_sizes, int n_in,
                              void* d_out, int out_size, void* d_ws,
                              size_t ws_size, hipStream_t stream) {
  (void)in_sizes; (void)n_in; (void)out_size; (void)ws_size;
  // setup_inputs() insertion order: x_seq, then per-layer {W_l,b_l,W_r,b_r,att,bias}, out_W, out_b
  const float* x_seq = (const float*)d_in[0];
  const float *W_l[3], *b_l[3], *W_r[3], *b_r[3], *attp[3], *biasp[3];
  for (int l = 0; l < 3; ++l) {
    const int b = 1 + 6 * l;
    W_l[l] = (const float*)d_in[b + 0];
    b_l[l] = (const float*)d_in[b + 1];
    W_r[l] = (const float*)d_in[b + 2];
    b_r[l] = (const float*)d_in[b + 3];
    attp[l]  = (const float*)d_in[b + 4];
    biasp[l] = (const float*)d_in[b + 5];
  }
  const float* out_W = (const float*)d_in[19];
  const float* out_b = (const float*)d_in[20];

  // carve workspace
  char* ws = (char*)d_ws;
  size_t off = 0;
  auto carve = [&](size_t bytes) -> void* {
    void* p = ws + off;
    off = (off + bytes + 255) & ~(size_t)255;
    return p;
  };
  const int dins[3] = {3, 64, 128};
  const int nouts[3] = {64, 128, 256};
  const int K32s[3] = {32, 64, 128};
  const int Cs[3] = {16, 32, 64};
  _Float16 *wtL[3], *wtR[3];
  for (int l = 0; l < 3; ++l) {
    wtL[l] = (_Float16*)carve((size_t)nouts[l] * K32s[l] * 2);
    wtR[l] = (_Float16*)carve((size_t)nouts[l] * K32s[l] * 2);
  }
  _Float16* wtO = (_Float16*)carve((size_t)NOUTF * KOUT * 2);
  _Float16* xb[2] = {(_Float16*)carve((size_t)MROW * 256 * 2),
                     (_Float16*)carve((size_t)MROW * 256 * 2)};
  _Float16* hlb = (_Float16*)carve((size_t)MROW * 256 * 2);
  _Float16* hrb = (_Float16*)carve((size_t)MROW * 256 * 2);

  // 1) input f32 -> f16
  {
    int n = MROW * 3;
    cvt_f32_to_f16_kernel<<<(n + 255) / 256, 256, 0, stream>>>(x_seq, xb[0], n);
  }
  // 2) weights: transpose + K-pad to f16
  for (int l = 0; l < 3; ++l) {
    int n = nouts[l] * K32s[l];
    cvt_w_transpose_kernel<<<(n + 255) / 256, 256, 0, stream>>>(
        W_l[l], wtL[l], dins[l], nouts[l], K32s[l]);
    cvt_w_transpose_kernel<<<(n + 255) / 256, 256, 0, stream>>>(
        W_r[l], wtR[l], dins[l], nouts[l], K32s[l]);
  }
  {
    int n = NOUTF * KOUT;  // 7,077,888
    cvt_w_transpose_kernel<<<(n + 255) / 256, 256, 0, stream>>>(
        out_W, wtO, KOUT, NOUTF, KOUT);
  }

  // 3) three GATv2 layers: proj (WMMA) then attention (VALU)
  const int attn_blocks = (MROW * 4) / 256;  // 2592, exact
  int cur = 0;
  for (int l = 0; l < 3; ++l) {
    dim3 pg(MROW / 128, nouts[l] / 64, 2);
    if (l == 0)
      proj_wmma_kernel<32><<<pg, 256, 0, stream>>>(
          xb[cur], dins[l], nouts[l], wtL[l], wtR[l], b_l[l], b_r[l], hlb, hrb);
    else if (l == 1)
      proj_wmma_kernel<64><<<pg, 256, 0, stream>>>(
          xb[cur], dins[l], nouts[l], wtL[l], wtR[l], b_l[l], b_r[l], hlb, hrb);
    else
      proj_wmma_kernel<128><<<pg, 256, 0, stream>>>(
          xb[cur], dins[l], nouts[l], wtL[l], wtR[l], b_l[l], b_r[l], hlb, hrb);
    gat_attn_kernel<<<attn_blocks, 256, 0, stream>>>(hlb, hrb, attp[l],
                                                     biasp[l], xb[1 - cur],
                                                     Cs[l]);
    cur = 1 - cur;
  }

  // 4) output head: (3072 x 13824) @ out_W + out_b -> d_out (f32)
  dim3 fg(NG / 128, NOUTF / 128);  // 24 x 4
  final_gemm_kernel<<<fg, 256, 0, stream>>>(xb[cur], wtO, out_b,
                                            (float*)d_out);
}
